// SpikePropLayer_34419867910580
// MI455X (gfx1250) — matrix operand order repaired and checked by reading, not compiled, not run on hardware
//
#include <hip/hip_runtime.h>

typedef __attribute__((ext_vector_type(16))) _Float16 v16h;
typedef __attribute__((ext_vector_type(8)))  _Float16 v8h;
typedef __attribute__((ext_vector_type(8)))  float    v8f;

#define B_    64
#define IN_   1024
#define OUT_  512
#define T_    512
#define TAU_  16.0f
#define VTH_  1.0f

// ---------------------------------------------------------------------------
// Init: out_spike = T (512.0f). atomicMin(int-bits) later finds first crossing.
// ---------------------------------------------------------------------------
__global__ void init_out_kernel(float* __restrict__ out) {
    int i = blockIdx.x * blockDim.x + threadIdx.x;
    if (i < B_ * OUT_) out[i] = (float)T_;
}

// ---------------------------------------------------------------------------
// Convert weights f32 -> f16 (row-major [OUT, IN]) into workspace.
// ---------------------------------------------------------------------------
__global__ void convert_w_kernel(const float* __restrict__ w, _Float16* __restrict__ wh) {
    int i = blockIdx.x * blockDim.x + threadIdx.x;
    if (i < OUT_ * IN_) wh[i] = (_Float16)w[i];
}

// ---------------------------------------------------------------------------
// Main kernel. One block = one (batch, 16-wide t-tile). 256 threads = 8 waves.
// Each wave owns 4 C-tiles (16 out x 16 t) -> block covers all 512 outputs,
// so the alpha-kernel values for this (b, t-tile) are computed exactly once.
// LDS holds the alpha B operands for ALL K (IN=1024), pre-swizzled into WMMA
// B-operand order: bop[kk][lane][j], K = kk*32 + (lane&16)/2 + (j&7) + (j&8)*2, N(t) = lane&15.
// Inner loop is ping-pong double-buffered so WMMAs for step kk overlap the
// operand loads for step kk+1.
// ---------------------------------------------------------------------------
template <bool W16>
__global__ __launch_bounds__(256) void spikeprop_kernel(
    const float* __restrict__ spike,      // [B, IN]
    const float* __restrict__ w32,        // [OUT, IN] f32
    const _Float16* __restrict__ w16,     // [OUT, IN] f16 (if W16)
    float* __restrict__ out)              // [B, OUT]
{
    __shared__ _Float16 bop[32 * 32 * 16];   // 32 KB: [kk][lane][j]

    const int b   = blockIdx.y;
    const int t0  = blockIdx.x * 16;
    const int tid = threadIdx.x;

    // ---- Phase 1: build alpha B-operand tiles (each thread: 4 k-rows x 16 t)
    for (int k = tid; k < IN_; k += 256) {
        const float s   = spike[b * IN_ + k];
        const int   kk  = k >> 5;
        const int   kr  = k & 31;
        const int   lb  = (kr & 8) << 1;     // lane group base (0 or 16)
        const int   j   = (kr & 7) | ((kr & 16) >> 1);
        _Float16* dst = &bop[(kk * 32 + lb) * 16 + j];
#pragma unroll
        for (int t = 0; t < 16; ++t) {
            const float tt = (float)(t0 + t) - s;
            float v = 0.0f;
            if (tt > 0.0f) {
                const float x = tt * (1.0f / TAU_);
                v = x * __expf(1.0f - x);
            }
            dst[t * 16] = (_Float16)v;       // lane = lb + t
        }
    }
    __syncthreads();

    // ---- Phase 2: WMMA GEMM over K (double-buffered)
    const int wave = tid >> 5;
    const int lane = tid & 31;
    const int mrow = lane & 15;              // A-row within tile
    const int kb   = (lane & 16) ? 8 : 0;    // A-operand K sub-base per lane group

    v8f acc[4] = {};
    const _Float16* pA16[4];
    const float*    pA32[4];
#pragma unroll
    for (int c = 0; c < 4; ++c) {
        const int row = (wave * 4 + c) * 16 + mrow;
        if (W16) pA16[c] = w16 + row * IN_ + kb;
        else     pA32[c] = w32 + row * IN_ + kb;
    }

    union AU { v16h v; v8h h[2]; };
    AU   aA[2][4];
    v16h bB[2];

    auto load_stage = [&](int kkIdx, int buf) {
        bB[buf] = *(const v16h*)&bop[(kkIdx * 32 + lane) * 16];
        const int k0 = kkIdx * 32;
#pragma unroll
        for (int c = 0; c < 4; ++c) {
            if (W16) {
                const _Float16* p = pA16[c] + k0;
                aA[buf][c].h[0] = *(const v8h*)(p);        // halves 0..7  : K kb+0..7
                aA[buf][c].h[1] = *(const v8h*)(p + 16);   // halves 8..15 : K kb+16..23
            } else {
                const float* p = pA32[c] + k0;
#pragma unroll
                for (int j = 0; j < 8; ++j) {
                    aA[buf][c].v[j]     = (_Float16)p[j];
                    aA[buf][c].v[j + 8] = (_Float16)p[16 + j];
                }
            }
        }
    };

    load_stage(0, 0);
    for (int kk = 0; kk < 32; kk += 2) {
        load_stage(kk + 1, 1);                 // prefetch odd step
#pragma unroll
        for (int c = 0; c < 4; ++c)
            acc[c] = __builtin_amdgcn_wmma_f32_16x16x32_f16(
                false, aA[0][c].v, false, bB[0], (short)0, acc[c], false, false);
        load_stage((kk + 2) & 31, 0);          // prefetch next even step (wraps harmlessly)
#pragma unroll
        for (int c = 0; c < 4; ++c)
            acc[c] = __builtin_amdgcn_wmma_f32_16x16x32_f16(
                false, aA[1][c].v, false, bB[1], (short)0, acc[c], false, false);
    }

    // ---- Phase 3: threshold -> first-crossing time via atomicMin on float bits
    const int ng  = lane & 15;                   // t within tile
    const int mhi = (lane & 16) ? 8 : 0;         // C layout: M = r + mhi
    const float tval  = (float)(t0 + ng);
    const int   tbits = __float_as_int(tval);    // non-negative: int order == float order
#pragma unroll
    for (int c = 0; c < 4; ++c) {
        const int obase = (wave * 4 + c) * 16 + mhi;
#pragma unroll
        for (int r = 0; r < 8; ++r) {
            if (acc[c][r] >= VTH_) {
                atomicMin((int*)&out[b * OUT_ + obase + r], tbits);
            }
        }
    }
}

// ---------------------------------------------------------------------------
extern "C" void kernel_launch(void* const* d_in, const int* in_sizes, int n_in,
                              void* d_out, int out_size, void* d_ws, size_t ws_size,
                              hipStream_t stream) {
    const float* spike  = (const float*)d_in[0];   // [B, IN]
    const float* weight = (const float*)d_in[1];   // [OUT, IN]
    float*       out    = (float*)d_out;           // [B, OUT]

    // init out_spike = T
    init_out_kernel<<<(B_ * OUT_ + 255) / 256, 256, 0, stream>>>(out);

    const size_t whBytes = (size_t)OUT_ * IN_ * sizeof(_Float16);
    dim3 grid(T_ / 16, B_);   // 32 t-tiles x 64 batches

    if (ws_size >= whBytes) {
        _Float16* wh = (_Float16*)d_ws;
        convert_w_kernel<<<(OUT_ * IN_ + 255) / 256, 256, 0, stream>>>(weight, wh);
        spikeprop_kernel<true><<<grid, 256, 0, stream>>>(spike, weight, wh, out);
    } else {
        spikeprop_kernel<false><<<grid, 256, 0, stream>>>(spike, weight, nullptr, out);
    }
}
